// SparseConv3D_77584289235263
// MI455X (gfx1250) — matrix operand order repaired and checked by reading, not compiled, not run on hardware
//
#include <hip/hip_runtime.h>

// ---------------------------------------------------------------------------
// SparseConv3D for MI455X (gfx1250, wave32, WMMA).
//
// Pipeline (all on `stream`, workspace-only scratch):
//  1. compute voxel index per point (int4)                       [16000 thr]
//  2. convert features fp32 -> bf16                              [2.05M thr]
//  3. convert+transpose weights fp32 [k][cin][cout] ->
//     bf16 [k][cout][cin] (so B fragments are contiguous loads)  [2.05M thr]
//  4. init neighbor table (-1) and per-tile offset masks (0)
//  5. brute-force neighbor build per batch (2000x2000 pairs):
//     writes nbr[i*125+k] = max point id in voxel idx_i+off_k,
//     and sets bit k in the 125-bit mask of tile i/16
//  6. WMMA kernel: 1000 blocks x 256 threads (8 waves).
//     Each block = 16 points; each wave = 16 output columns.
//     Loop only over offsets whose mask bit is set (~17 of 125),
//     gather 16x128 bf16 A-tile to LDS, 4x v_wmma_f32_16x16x32_bf16
//     per wave per offset, f32 accumulation, direct f32 stores.
// ---------------------------------------------------------------------------

typedef __attribute__((ext_vector_type(16))) __bf16 v16bf;
typedef __attribute__((ext_vector_type(8)))  __bf16 v8bf;
typedef __attribute__((ext_vector_type(8)))  float  v8f;

#define N_PTS   16000
#define G_PER_B 2000
#define CH      128
#define KOFF    125
#define TILE    16
#define NTILES  (N_PTS / TILE)     // 1000
#define APAD    136                // 128 + 8 bf16 pad (LDS bank spread)

// ---- workspace layout (bytes, all multiples of 256) ----
#define WS_IDX4   0                           // 16000 * 16      = 256000
#define WS_NBR    256000                      // 16000*125*4     = 8000000
#define WS_FEATB  8256000                     // 16000*128*2     = 4096000
#define WS_WT     12352000                    // 125*128*128*2   = 4096000
#define WS_MASK   16448000                    // 1000*4*4        = 16000
// total: 16464000 bytes

__global__ void compute_idx_kernel(const float* __restrict__ anchor,
                                   int4* __restrict__ idx4, int n) {
  int i = blockIdx.x * blockDim.x + threadIdx.x;
  if (i >= n) return;
  const float lo0 = -20.0f, lo1 = -20.0f, lo2 = -2.3f;
  const float r0 = 40.0f, r1 = 40.0f, r2 = 3.2f;
  float x = anchor[i * 3 + 0] * r0 + lo0;
  float y = anchor[i * 3 + 1] * r1 + lo1;
  float z = anchor[i * 3 + 2] * r2 + lo2;
  int ix = (int)((x - lo0) / 0.1f);
  int iy = (int)((y - lo1) / 0.1f);
  int iz = (int)((z - lo2) / 0.1f);
  idx4[i] = make_int4(ix, iy, iz, 0);
}

__global__ void convert_feat_kernel(const float* __restrict__ f,
                                    __bf16* __restrict__ fb, int n) {
  int i = blockIdx.x * blockDim.x + threadIdx.x;
  if (i < n) fb[i] = (__bf16)f[i];
}

// w: [125][cin=128][cout=128] fp32  ->  wT: [125][cout=128][cin=128] bf16
__global__ void convert_w_kernel(const float* __restrict__ w,
                                 __bf16* __restrict__ wT) {
  int t = blockIdx.x * blockDim.x + threadIdx.x;
  if (t >= KOFF * CH * CH) return;
  int off  = t / (CH * CH);
  int rem  = t % (CH * CH);
  int cout = rem / CH;
  int cin  = rem % CH;
  wT[t] = (__bf16)w[off * CH * CH + cin * CH + cout];
}

__global__ void init_kernel(int* __restrict__ nbr, unsigned* __restrict__ mask) {
  int t = blockIdx.x * blockDim.x + threadIdx.x;
  if (t < N_PTS * KOFF) nbr[t] = -1;
  else if (t < N_PTS * KOFF + NTILES * 4) mask[t - N_PTS * KOFF] = 0u;
}

// one block per point i; threads scan the 2000 points of i's batch
__global__ __launch_bounds__(128) void build_nbr_kernel(
    const int4* __restrict__ idx4, int* __restrict__ nbr,
    unsigned* __restrict__ mask) {
  int i = blockIdx.x;                 // global point id
  int base = (i / G_PER_B) * G_PER_B; // batch start
  int4 pi = idx4[i];
  for (int jl = threadIdx.x; jl < G_PER_B; jl += 128) {
    int j = base + jl;
    int4 pj = idx4[j];
    int dx = pj.x - pi.x, dy = pj.y - pi.y, dz = pj.z - pi.z;
    if (dx >= -2 && dx <= 2 && dy >= -2 && dy <= 2 && dz >= -2 && dz <= 2) {
      int k = (dx + 2) * 25 + (dy + 2) * 5 + (dz + 2);
      atomicMax(&nbr[i * KOFF + k], j);
      atomicOr(&mask[(i >> 4) * 4 + (k >> 5)], 1u << (k & 31));
    }
  }
}

__global__ __launch_bounds__(256) void conv_wmma_kernel(
    const int* __restrict__ nbr, const __bf16* __restrict__ featb,
    const __bf16* __restrict__ wT, const unsigned* __restrict__ mask,
    float* __restrict__ out) {
  __shared__ __bf16 As[TILE * APAD];  // 16 x (128+8) bf16 = 4352 B

  int tile = blockIdx.x;
  int tid  = threadIdx.x;
  int lane = tid & 31;
  int wave = tid >> 5;                // 0..7 -> N-tile (16 output cols)
  int row0 = tile * TILE;

  int gr = tid >> 4;                  // gather row 0..15
  int gc = (tid & 15) * 8;            // gather col chunk (8 bf16 = 16 B)

  int m    = lane & 15;               // A row / B column within tile
  int selA = (lane < 16) ? 0 : 8;     // A chunk select per 16-bit A layout
  int selB = (lane < 16) ? 0 : 16;    // K-half offset; unused (B loads use selA)

  unsigned mk0 = mask[tile * 4 + 0];
  unsigned mk1 = mask[tile * 4 + 1];
  unsigned mk2 = mask[tile * 4 + 2];
  unsigned mk3 = mask[tile * 4 + 3];

  v8f acc = {};
  const __bf16* wcol = wT + (wave * 16 + m) * CH;

  for (int k = 0; k < KOFF; ++k) {
    unsigned mw = (k < 32) ? mk0 : (k < 64) ? mk1 : (k < 96) ? mk2 : mk3;
    if (!((mw >> (k & 31)) & 1u)) continue;   // block-uniform skip

    __syncthreads();  // protect As against previous iteration's readers
    int j = nbr[(row0 + gr) * KOFF + k];
    v8bf v = {};
    if (j >= 0) v = *(const v8bf*)(featb + j * CH + gc);
    *(v8bf*)(As + gr * APAD + gc) = v;
    __syncthreads();

    const __bf16* wk = wcol + k * CH * CH;
#pragma unroll
    for (int kk = 0; kk < 4; ++kk) {
      // A: lanes<16 hold K {0..7,16..23}, lanes>=16 hold K {8..15,24..31}
      v8bf a0 = *(const v8bf*)(As + m * APAD + kk * 32 + selA);
      v8bf a1 = *(const v8bf*)(As + m * APAD + kk * 32 + selA + 16);
      v16bf a = __builtin_shufflevector(a0, a1, 0, 1, 2, 3, 4, 5, 6, 7,
                                        8, 9, 10, 11, 12, 13, 14, 15);
      // B: lanes<16 hold K {0..7,16..23}, lanes>=16 hold K {8..15,24..31} (wT is [k][cout][cin])
      v8bf b0 = *(const v8bf*)(wk + kk * 32 + selA);
      v8bf b1 = *(const v8bf*)(wk + kk * 32 + selA + 16);
      v16bf b = __builtin_shufflevector(b0, b1, 0, 1, 2, 3, 4, 5, 6, 7,
                                        8, 9, 10, 11, 12, 13, 14, 15);
      acc = __builtin_amdgcn_wmma_f32_16x16x32_bf16(
          false, a, false, b, (short)0, acc, false, false);
    }
  }

  // C/D layout: VGPR r -> M=r (lanes 0-15) / M=r+8 (lanes 16-31), N=lane&15
  int col   = wave * 16 + m;
  int rbase = row0 + ((lane < 16) ? 0 : 8);
#pragma unroll
  for (int r = 0; r < 8; ++r)
    out[(rbase + r) * CH + col] = acc[r];
}

extern "C" void kernel_launch(void* const* d_in, const int* in_sizes, int n_in,
                              void* d_out, int out_size, void* d_ws,
                              size_t ws_size, hipStream_t stream) {
  (void)in_sizes; (void)n_in; (void)out_size; (void)ws_size;
  const float* feat   = (const float*)d_in[0];  // (8,2000,128) f32
  const float* anchor = (const float*)d_in[1];  // (8,2000,3)   f32
  const float* w      = (const float*)d_in[2];  // (125,128,128) f32
  float* out = (float*)d_out;                   // (8,2000,128) f32

  char* ws = (char*)d_ws;
  int4*     idx4  = (int4*)(ws + WS_IDX4);
  int*      nbr   = (int*)(ws + WS_NBR);
  __bf16*   featb = (__bf16*)(ws + WS_FEATB);
  __bf16*   wT    = (__bf16*)(ws + WS_WT);
  unsigned* mask  = (unsigned*)(ws + WS_MASK);

  compute_idx_kernel<<<(N_PTS + 255) / 256, 256, 0, stream>>>(anchor, idx4, N_PTS);
  convert_feat_kernel<<<(N_PTS * CH + 255) / 256, 256, 0, stream>>>(feat, featb, N_PTS * CH);
  convert_w_kernel<<<(KOFF * CH * CH + 255) / 256, 256, 0, stream>>>(w, wT);
  init_kernel<<<(N_PTS * KOFF + NTILES * 4 + 255) / 256, 256, 0, stream>>>(nbr, mask);
  build_nbr_kernel<<<N_PTS, 128, 0, stream>>>(idx4, nbr, mask);
  conv_wmma_kernel<<<NTILES, 256, 0, stream>>>(nbr, featb, wT, mask, out);
}
